// HiveGNNPolicyHetero_82764019794262
// MI455X (gfx1250) — matrix operand order repaired and checked by reading, not compiled, not run on hardware
//
#include <hip/hip_runtime.h>
#include <math.h>

#define HID 256

typedef __attribute__((ext_vector_type(16))) __bf16 v16bf;
typedef __attribute__((ext_vector_type(8)))  float  v8f;
typedef __attribute__((ext_vector_type(4)))  unsigned int u32x4;
typedef __attribute__((ext_vector_type(8)))  int i32x8;
typedef __attribute__((ext_vector_type(4)))  int i32x4;

#if __has_builtin(__builtin_amdgcn_tensor_load_to_lds)
#define GEMM_TDM 1
#endif

__device__ inline unsigned short f2bf(float f) {
  unsigned int x = __float_as_uint(f);
  unsigned int r = x + 0x7fffu + ((x >> 16) & 1u);   // round-to-nearest-even
  return (unsigned short)(r >> 16);
}

__device__ inline unsigned int pk2bf(float lo, float hi) {
#if __has_builtin(__builtin_amdgcn_cvt_pk_bf16_f32)
  typedef __attribute__((ext_vector_type(2))) __bf16 v2bf;
  v2bf r = __builtin_amdgcn_cvt_pk_bf16_f32(lo, hi);
  return __builtin_bit_cast(unsigned int, r);
#else
  return (unsigned int)f2bf(lo) | ((unsigned int)f2bf(hi) << 16);
#endif
}

__device__ inline void waitTensor0() {
#if __has_builtin(__builtin_amdgcn_s_wait_tensorcnt)
  __builtin_amdgcn_s_wait_tensorcnt((short)0);
#else
  asm volatile("s_wait_tensorcnt 0x0" ::: "memory");
#endif
}

// ---------------------------------------------------------------------------
// Weight pre-swizzle: f32 [K,N] -> bf16 fragments for the WMMA B operand (contiguous k order).
// Fragment (kstep s, ntile t): 32 lanes x 16 elems; lane = col + 16*(k>=16),
// elem e -> k = s*32 + 16*(lane>>4) + e, n = t*16 + (lane&15). Zero padded.
// ---------------------------------------------------------------------------
__global__ void k_swz(const float* __restrict__ W, int K, int N, int ntl,
                      unsigned short* __restrict__ dst, int total)
{
  int i = blockIdx.x * 256 + threadIdx.x;
  if (i >= total) return;
  int e    = i & 15;
  int lane = (i >> 4) & 31;
  int t    = (i >> 9) % ntl;
  int s    = i / (ntl * 512);
  int k = s * 32 + 16 * (lane >> 4) + e;
  int n = t * 16 + (lane & 15);
  float f = (k < K && n < N) ? W[(size_t)k * N + n] : 0.0f;
  dst[i] = f2bf(f);
}

// ---------------------------------------------------------------------------
// WMMA GEMM: C[M,N] = act(A[M,K] @ W + bias), W pre-swizzled bf16.
// Block tile 64x128: 8 wave32 waves = 4 mtiles x 2 nhalves, 4 accum each
// (4 unconditional v_wmma per K-step; OOB n-tiles clamp their *load* index
// and are discarded by the store mask -> EXEC stays all-1s, no phi copies).
// A tile (64x32 f32) double-buffered in LDS via the Tensor Data Mover,
// overlapped with compute; B fragments stream from global swizzled layout
// with coalesced 32B loads directly into the WMMA operands.
// ---------------------------------------------------------------------------
__global__ void __launch_bounds__(256)
k_gemm(const float* __restrict__ A, const unsigned short* __restrict__ Bswz,
       const float* __restrict__ bias, float* __restrict__ C,
       int M, int N, int K, int act)
{
  __shared__ float tile[2][64][32];            // 2 x 8KB
  const int tid  = threadIdx.x;
  const int lane = tid & 31;
  const int wave = tid >> 5;
  const int mt   = wave >> 1;                  // 0..3
  const int nh   = wave & 1;                   // 0..1
  const int m0   = blockIdx.x * 64;
  const int ntl  = (N + 15) >> 4;              // fragments per kstep in Bswz
  const int ksteps = (K + 31) >> 5;

  v8f acc[4] = {};

  // --- consume one staged A tile: build A fragment, 4x (B load + WMMA) ---
  auto consume = [&](int s, int buf) {
    const int arow = mt * 16 + (lane & 15);
    const int half = lane >> 4;
    const float* rp = &tile[buf][arow][0];
    float4 q0 = *(const float4*)(rp + 16 * half);
    float4 q1 = *(const float4*)(rp + 16 * half + 4);
    float4 q2 = *(const float4*)(rp + 16 * half + 8);
    float4 q3 = *(const float4*)(rp + 16 * half + 12);
    union { unsigned int u[8]; v16bf v; } fa;
    fa.u[0] = pk2bf(q0.x, q0.y); fa.u[1] = pk2bf(q0.z, q0.w);
    fa.u[2] = pk2bf(q1.x, q1.y); fa.u[3] = pk2bf(q1.z, q1.w);
    fa.u[4] = pk2bf(q2.x, q2.y); fa.u[5] = pk2bf(q2.z, q2.w);
    fa.u[6] = pk2bf(q3.x, q3.y); fa.u[7] = pk2bf(q3.z, q3.w);
    #pragma unroll
    for (int t = 0; t < 4; ++t) {
      int ntg = blockIdx.y * 8 + nh * 4 + t;
      if (ntg > ntl - 1) ntg = ntl - 1;        // clamp (no divergent branch)
      size_t foff = ((size_t)(s * ntl + ntg) * 32 + lane) * 16;
      v16bf fb = *(const v16bf*)(Bswz + foff); // 32B coalesced
      acc[t] = __builtin_amdgcn_wmma_f32_16x16x32_bf16(
          false, fa.v, false, fb, (short)0, acc[t], false, false);
    }
  };

#ifdef GEMM_TDM
  // --- TDM issue: 2D tile descriptor per ISA 8.3/8.4 (OOB reads -> 0) ---
  auto tdmIssue = [&](int s, int buf) {
    const int k0 = s * 32;
    unsigned long long ga =
        (unsigned long long)(uintptr_t)(A + ((size_t)m0 * K + k0));
    unsigned int ldsoff = (unsigned int)(uintptr_t)&tile[buf][0][0];
    unsigned int td0 = (unsigned int)(K - k0);                  // cols left
    int mrem = M - m0; if (mrem < 1) mrem = 1;
    unsigned int td1 = (unsigned int)mrem;                      // rows left
    unsigned long long st0 = (unsigned long long)(unsigned int)K;

    u32x4 g0;
    g0.x = 1u;                                                  // count=1
    g0.y = ldsoff;                                              // lds_addr
    g0.z = (unsigned int)ga;                                    // gaddr lo
    g0.w = ((unsigned int)(ga >> 32) & 0x01FFFFFFu) | (2u << 30); // hi|type=2
    i32x8 g1;
    g1[0] = (int)(2u << 16);                                    // data_size=4B
    g1[1] = (int)((td0 & 0xFFFFu) << 16);                       // dim0 lo
    g1[2] = (int)((td0 >> 16) | ((td1 & 0xFFFFu) << 16));       // dim0 hi|dim1 lo
    g1[3] = (int)((td1 >> 16) | (32u << 16));                   // dim1 hi|tile0=32
    g1[4] = (int)64u;                                           // tile1=64
    g1[5] = (int)(unsigned int)(st0 & 0xFFFFFFFFu);             // stride0 lo
    g1[6] = (int)((unsigned int)(st0 >> 32) & 0xFFFFu);         // stride0 hi
    g1[7] = 0;
    i32x4 g2 = {0, 0, 0, 0}, g3 = {0, 0, 0, 0};
    i32x8 g4 = {0, 0, 0, 0, 0, 0, 0, 0};
    __builtin_amdgcn_tensor_load_to_lds(g0, g1, g2, g3, g4, 0);
  };

  // --- software pipeline: DMA of tile s+1 overlaps compute of tile s ---
  if (wave == 0) { tdmIssue(0, 0); waitTensor0(); }
  __syncthreads();
  for (int s = 0; s < ksteps; ++s) {
    const int cur = s & 1;
    if (wave == 0 && s + 1 < ksteps) tdmIssue(s + 1, 1 - cur);
    consume(s, cur);
    if (wave == 0 && s + 1 < ksteps) waitTensor0();
    __syncthreads();
  }
#else
  // --- fallback: branch-free clamped copy, 2x16B per thread ---
  for (int s = 0; s < ksteps; ++s) {
    const int k0 = s * 32;
    const size_t lastOff = (size_t)M * K * 4 - 16;
    #pragma unroll
    for (int cpt = 0; cpt < 2; ++cpt) {
      int chunk = tid * 2 + cpt;           // 0..511 (64 rows x 8 chunks)
      int i  = chunk >> 3;
      int c8 = chunk & 7;
      int gm = m0 + i; if (gm > M - 1) gm = M - 1;
      size_t boff = ((size_t)gm * K + k0) * 4 + (size_t)c8 * 16;
      if (boff > lastOff) boff = lastOff;
      const float* sp = (const float*)((const char*)A + boff);
      float* dp = &tile[0][i][c8 * 4];
      dp[0] = sp[0]; dp[1] = sp[1]; dp[2] = sp[2]; dp[3] = sp[3];
    }
    __syncthreads();
    consume(s, 0);
    __syncthreads();
  }
#endif

  // ---- store C (C/D layout: VGPR r, lane L -> M=r+8*(L>>4), N=L&15) ----
  const int rowHalf = (lane >> 4) * 8;
  const int col     = lane & 15;
  #pragma unroll
  for (int t = 0; t < 4; ++t) {
    int gn = blockIdx.y * 128 + (nh * 4 + t) * 16 + col;   // unclamped index
    #pragma unroll
    for (int r = 0; r < 8; ++r) {
      int gm = m0 + mt * 16 + rowHalf + r;
      if (gm < M && gn < N) {
        float v = acc[t][r] + (bias ? bias[gn] : 0.0f);
        if (act == 1)      v = fmaxf(v, 0.0f);
        else if (act == 2) v = tanhf(v);
        C[(size_t)gm * N + gn] = v;
      }
    }
  }
}

// ---------------------------------------------------------------------------
// elementwise / edge kernels
// ---------------------------------------------------------------------------
__device__ inline void atomicMaxF(float* addr, float val) {
  if (val >= 0.0f) atomicMax((int*)addr, __float_as_int(val));
  else             atomicMin((unsigned int*)addr, __float_as_uint(val));
}

__global__ void k_fill(float* __restrict__ p, float v, int n) {
  int i = blockIdx.x * blockDim.x + threadIdx.x;
  if (i < n) p[i] = v;
}

__global__ void k_gat_logits(const float* __restrict__ XL, const float* __restrict__ XR,
                             const int* __restrict__ ei, int E,
                             const float* __restrict__ att,
                             const float* __restrict__ ea,
                             const float* __restrict__ u, const float* __restrict__ v,
                             float* __restrict__ logit, float* __restrict__ segmax)
{
  int t = blockIdx.x * blockDim.x + threadIdx.x;
  if (t >= E * 4) return;
  int e = t >> 2, h = t & 3;
  int src = ei[e], dst = ei[E + e];
  const float* xl = XL + (size_t)src * HID + h * 64;
  const float* xr = XR + (size_t)dst * HID + h * 64;
  const float* at = att + h * 64;
  float s = 0.0f;
  if (ea) {
    float eaw = ea[e];
    const float* up = u + h * 64; const float* vp = v + h * 64;
    for (int d = 0; d < 64; ++d) {
      float m = xl[d] + xr[d] + eaw * up[d] + vp[d];
      m = (m > 0.0f) ? m : 0.2f * m;
      s += m * at[d];
    }
  } else {
    for (int d = 0; d < 64; ++d) {
      float m = xl[d] + xr[d];
      m = (m > 0.0f) ? m : 0.2f * m;
      s += m * at[d];
    }
  }
  logit[t] = s;
  atomicMaxF(&segmax[dst * 4 + h], s);
}

__global__ void k_gat_scatter(const float* __restrict__ XL, const int* __restrict__ ei, int E,
                              const float* __restrict__ logit, const float* __restrict__ segmax,
                              float* __restrict__ denom, float* __restrict__ gout)
{
  int e = blockIdx.x, c = threadIdx.x, h = c >> 6;
  int src = ei[e], dst = ei[E + e];
  float lm = segmax[dst * 4 + h];
  if (!(lm > -1e37f)) lm = 0.0f;
  float ex = __expf(logit[e * 4 + h] - lm);
  if ((c & 63) == 0) atomicAdd(&denom[dst * 4 + h], ex);
  atomicAdd(&gout[(size_t)dst * HID + c], ex * XL[(size_t)src * HID + c]);
}

__global__ void k_gat_final(const float* __restrict__ gout, const float* __restrict__ denom,
                            const float* __restrict__ bias, float* __restrict__ acc)
{
  int i = blockIdx.x, c = threadIdx.x, h = c >> 6;
  acc[(size_t)i * HID + c] += gout[(size_t)i * HID + c] / (denom[i * 4 + h] + 1e-16f) + bias[c];
}

__global__ void k_post(float* __restrict__ x, const float* __restrict__ acc,
                       const float* __restrict__ g, const float* __restrict__ b,
                       const float* __restrict__ m, const float* __restrict__ v)
{
  int i = blockIdx.x, c = threadIdx.x;
  size_t o = (size_t)i * HID + c;
  float a = fmaxf(acc[o], 0.0f) + x[o];
  x[o] = g[c] * (a - m[c]) * rsqrtf(v[c] + 1e-5f) + b[c];
}

__global__ void k_uv(const float* __restrict__ Wm, const float* __restrict__ bm,
                     const float* __restrict__ We, float* __restrict__ u, float* __restrict__ v)
{
  int t = blockIdx.x * 256 + threadIdx.x;
  if (t < 256) {
    float s = 0.0f;
    for (int k = 0; k < 256; ++k) s += Wm[k] * We[k * 256 + t];
    u[t] = s;
  } else if (t < 512) {
    int c = t - 256; float s = 0.0f;
    for (int k = 0; k < 256; ++k) s += bm[k] * We[k * 256 + c];
    v[c] = s;
  }
}

__global__ void k_feats(const float* __restrict__ xip, const float* __restrict__ xoop,
                        const float* __restrict__ xd,
                        const int* __restrict__ ei_mid, const int* __restrict__ ei_mod,
                        float* __restrict__ feats)
{
  int e = blockIdx.x, c = threadIdx.x;
  float a, b;
  if (e < 30000) {
    a = xip[(size_t)ei_mid[e] * HID + c];
    b = xd [(size_t)ei_mid[30000 + e] * HID + c];
  } else {
    int e2 = e - 30000;
    a = xoop[(size_t)ei_mod[e2] * HID + c];
    b = xd  [(size_t)ei_mod[10000 + e2] * HID + c];
  }
  feats[(size_t)e * HID + c] = 0.5f * (a + b);
}

__global__ void k_pool(const float* __restrict__ x, int nrows,
                       float* __restrict__ out, float invn)
{
  int c = threadIdx.x;
  int r0 = blockIdx.x * 256;
  int r1 = min(r0 + 256, nrows);
  float s = 0.0f;
  for (int r = r0; r < r1; ++r) s += x[(size_t)r * HID + c];
  atomicAdd(&out[c], s * invn);
}

__global__ void k_scatter_logits(const float* __restrict__ plog,
                                 const float* __restrict__ ea_mid,
                                 const float* __restrict__ ea_mod,
                                 const int* __restrict__ mtai, float* __restrict__ out)
{
  int e = blockIdx.x * blockDim.x + threadIdx.x;
  if (e >= 40000) return;
  float attr = (e < 30000) ? ea_mid[e] : ea_mod[e - 30000];
  int idx = mtai[e];
  if (attr == 1.0f && idx >= 0 && idx < 50000) out[idx] = plog[e];
}

// ---------------------------------------------------------------------------
// host orchestration
// ---------------------------------------------------------------------------
struct Lin { const float *W, *b; };
struct Gat { const float *Wl,*bl,*Wr,*br,*att,*bias,*We; };
struct Bnp { const float *g,*b,*m,*v; };
struct SwzW { const unsigned short* p; };

extern "C" void kernel_launch(void* const* d_in, const int* in_sizes, int n_in,
                              void* d_out, int out_size, void* d_ws, size_t ws_size,
                              hipStream_t stream)
{
  (void)in_sizes; (void)n_in; (void)out_size; (void)ws_size;

  const float* x_ip  = (const float*)d_in[0];
  const float* x_oop = (const float*)d_in[1];
  const float* x_d   = (const float*)d_in[2];
  const int* ei_nn  = (const int*)d_in[3];
  const int* ei_nd  = (const int*)d_in[4];
  const int* ei_dn  = (const int*)d_in[5];
  const int* ei_dd  = (const int*)d_in[6];
  const int* ei_mid = (const int*)d_in[7];
  const int* ei_mod = (const int*)d_in[8];
  const float* ea_mid = (const float*)d_in[9];
  const float* ea_mod = (const float*)d_in[10];
  const int* mtai = (const int*)d_in[11];

  // params: depth-first in make_params() insertion order
  int pi = 12;
  auto nf  = [&]() { return (const float*)d_in[pi++]; };
  auto lin = [&]() { Lin l; l.W = nf(); l.b = nf(); return l; };
  auto gat = [&](bool edge) {
    Gat g; g.Wl=nf(); g.bl=nf(); g.Wr=nf(); g.br=nf(); g.att=nf(); g.bias=nf();
    g.We = edge ? nf() : nullptr; return g;
  };
  auto bnp = [&]() { Bnp b; b.g=nf(); b.b=nf(); b.m=nf(); b.v=nf(); return b; };

  Lin emb_ip = lin(), emb_oop = lin(), emb_dest = lin(), emb_move = lin();
  Gat Lcv[4][6];   // nn, nd, dn, dd, mid, mod
  Bnp Bip[4], Bd[4];
  for (int l = 0; l < 4; ++l) {
    Lcv[l][0]=gat(false); Lcv[l][1]=gat(false); Lcv[l][2]=gat(false); Lcv[l][3]=gat(false);
    Lcv[l][4]=gat(true);  Lcv[l][5]=gat(true);  Bip[l]=bnp(); Bd[l]=bnp();
  }
  Lin pol[3] = {lin(), lin(), lin()};
  Lin val[3] = {lin(), lin(), lin()};

  // ---- workspace carve ----
  float* w = (float*)d_ws;  size_t off = 0;
  auto alloc = [&](size_t n) { float* p = w + off; off += n; return p; };
  float* xip   = alloc(20000ull * 256);
  float* xoop  = alloc( 5000ull * 256);
  float* xd    = alloc(30000ull * 256);
  float* XL    = alloc(30000ull * 256);   // + XR contiguous: reused as feats
  float* XR    = alloc(30000ull * 256);
  float* accip = alloc(20000ull * 256);   // reused as p1
  float* accd  = alloc(30000ull * 256);   // reused as p2
  float* gout  = alloc(30000ull * 256);   // reused as plog
  float* logit = alloc(40000ull * 4);
  float* segmx = alloc(30000ull * 4);
  float* denom = alloc(30000ull * 4);
  float* umid  = alloc(256); float* vmid = alloc(256);
  float* umod  = alloc(256); float* vmod = alloc(256);
  float* pooled= alloc(768);
  float* v1    = alloc(256); float* v2 = alloc(128);
  (void)XR;
  float* feats = XL;      // 40000*256 fits in XL+XR
  float* p1    = accip;   // 40000*128
  float* p2    = accd;    // 40000*64
  float* plog  = gout;    // 40000

  // bf16 swizzled weight region (after f32 region; 32B aligned by construction)
  unsigned short* swzCur = (unsigned short*)(w + off);

  auto prep = [&](const float* W, int K, int N) {
    int ks = (K + 31) / 32, ntl = (N + 15) / 16;
    size_t cnt = (size_t)ks * ntl * 512;
    unsigned short* dst = swzCur; swzCur += cnt;
    k_swz<<<(int)((cnt + 255) / 256), 256, 0, stream>>>(W, K, N, ntl, dst, (int)cnt);
    SwzW s{dst}; return s;
  };

  // pre-swizzle every GEMM weight once
  SwzW sw_emb_ip = prep(emb_ip.W, 10, 256);
  SwzW sw_emb_oop = prep(emb_oop.W, 10, 256);
  SwzW sw_emb_dest = prep(emb_dest.W, 10, 256);
  SwzW swl[4][6], swr[4][6];
  for (int l = 0; l < 4; ++l)
    for (int c = 0; c < 6; ++c) {
      swl[l][c] = prep(Lcv[l][c].Wl, 256, 256);
      swr[l][c] = prep(Lcv[l][c].Wr, 256, 256);
    }
  SwzW sw_pol[3] = {prep(pol[0].W, 256, 128), prep(pol[1].W, 128, 64), prep(pol[2].W, 64, 1)};
  SwzW sw_val[3] = {prep(val[0].W, 768, 256), prep(val[1].W, 256, 128), prep(val[2].W, 128, 1)};

  auto gemm = [&](const float* A, SwzW sw, const float* bias, float* C,
                  int M, int N, int K, int act) {
    dim3 g((M + 63) / 64, (N + 127) / 128);
    k_gemm<<<g, 256, 0, stream>>>(A, sw.p, bias, C, M, N, K, act);
  };
  auto fill = [&](float* p, float vv, size_t n) {
    int nn = (int)n;
    k_fill<<<(nn + 255) / 256, 256, 0, stream>>>(p, vv, nn);
  };
  auto conv = [&](const float* xs, int ns, const float* xdst, int nd, const int* ei, int E,
                  const Gat& G, SwzW sl, SwzW sr,
                  const float* ea, const float* uu, const float* vv, float* acc) {
    gemm(xs,   sl, G.bl, XL, ns, 256, 256, 0);
    gemm(xdst, sr, G.br, XR, nd, 256, 256, 0);
    fill(segmx, -INFINITY, (size_t)nd * 4);
    fill(denom, 0.0f, (size_t)nd * 4);
    fill(gout,  0.0f, (size_t)nd * 256);
    k_gat_logits<<<(E * 4 + 255) / 256, 256, 0, stream>>>(XL, XR, ei, E, G.att, ea, uu, vv, logit, segmx);
    k_gat_scatter<<<E, 256, 0, stream>>>(XL, ei, E, logit, segmx, denom, gout);
    k_gat_final<<<nd, 256, 0, stream>>>(gout, denom, G.bias, acc);
  };

  // 1) embeddings (K=10, zero-padded via swizzled B)
  gemm(x_ip,  sw_emb_ip,   emb_ip.b,   xip,  20000, 256, 10, 1);
  gemm(x_oop, sw_emb_oop,  emb_oop.b,  xoop,  5000, 256, 10, 1);
  gemm(x_d,   sw_emb_dest, emb_dest.b, xd,   30000, 256, 10, 1);

  // 2) GATv2 layers
  for (int l = 0; l < 4; ++l) {
    k_uv<<<2, 256, 0, stream>>>(emb_move.W, emb_move.b, Lcv[l][4].We, umid, vmid);
    k_uv<<<2, 256, 0, stream>>>(emb_move.W, emb_move.b, Lcv[l][5].We, umod, vmod);
    fill(accip, 0.0f, 20000ull * 256);
    fill(accd,  0.0f, 30000ull * 256);
    conv(xip, 20000, xip, 20000, ei_nn, 40000, Lcv[l][0], swl[l][0], swr[l][0], nullptr, nullptr, nullptr, accip);
    conv(xd,  30000, xip, 20000, ei_dn, 40000, Lcv[l][2], swl[l][2], swr[l][2], nullptr, nullptr, nullptr, accip);
    conv(xip, 20000, xd,  30000, ei_nd, 40000, Lcv[l][1], swl[l][1], swr[l][1], nullptr, nullptr, nullptr, accd);
    conv(xd,  30000, xd,  30000, ei_dd, 40000, Lcv[l][3], swl[l][3], swr[l][3], nullptr, nullptr, nullptr, accd);
    conv(xip, 20000, xd,  30000, ei_mid, 30000, Lcv[l][4], swl[l][4], swr[l][4], ea_mid, umid, vmid, accd);
    conv(xoop, 5000, xd,  30000, ei_mod, 10000, Lcv[l][5], swl[l][5], swr[l][5], ea_mod, umod, vmod, accd);
    k_post<<<20000, 256, 0, stream>>>(xip, accip, Bip[l].g, Bip[l].b, Bip[l].m, Bip[l].v);
    k_post<<<30000, 256, 0, stream>>>(xd,  accd,  Bd[l].g,  Bd[l].b,  Bd[l].m,  Bd[l].v);
  }

  // 3) policy head
  k_feats<<<40000, 256, 0, stream>>>(xip, xoop, xd, ei_mid, ei_mod, feats);
  gemm(feats, sw_pol[0], pol[0].b, p1,   40000, 128, 256, 1);
  gemm(p1,    sw_pol[1], pol[1].b, p2,   40000,  64, 128, 1);
  gemm(p2,    sw_pol[2], pol[2].b, plog, 40000,   1,  64, 0);

  // 4) value head
  fill(pooled, 0.0f, 768);
  k_pool<<<(20000 + 255) / 256, 256, 0, stream>>>(xip,  20000, pooled,       1.0f / 20000.0f);
  k_pool<<<( 5000 + 255) / 256, 256, 0, stream>>>(xoop,  5000, pooled + 256, 1.0f /  5000.0f);
  k_pool<<<(30000 + 255) / 256, 256, 0, stream>>>(xd,   30000, pooled + 512, 1.0f / 30000.0f);
  gemm(pooled, sw_val[0], val[0].b, v1, 1, 256, 768, 1);
  gemm(v1,     sw_val[1], val[1].b, v2, 1, 128, 256, 1);
  gemm(v2,     sw_val[2], val[2].b, ((float*)d_out) + 50000, 1, 1, 128, 2);

  // 5) action logits
  fill((float*)d_out, -INFINITY, 50000);
  k_scatter_logits<<<(40000 + 255) / 256, 256, 0, stream>>>(plog, ea_mid, ea_mod, mtai, (float*)d_out);
}
